// Attention1D_15831249453073
// MI455X (gfx1250) — matrix operand order repaired and checked by reading, not compiled, not run on hardware
//
#include <hip/hip_runtime.h>

#define BB 4
#define CC 128
#define HH 64
#define WW 256
#define RR 8
#define WP (WW + 2 * RR)  /* 272 padded x extent */
#define FP (WP + 8)       /* 280: fv row stride (bank-skewed) */
#define CP (CC + 8)       /* 136: channel-major row stride */
#define PP 40             /* PT row stride (32 + 8 pad) */
#define NTHREADS 256
#define NWAVES 8

typedef __attribute__((ext_vector_type(16))) __bf16 v16bf;
typedef __attribute__((ext_vector_type(8))) __bf16 v8bf;
typedef __attribute__((ext_vector_type(8))) float v8f;

__device__ __forceinline__ v8f wmma_bf16(const v16bf& a, const v16bf& b, const v8f& c) {
  // D(16x16 f32) = A(16x32 bf16) * B(32x16 bf16) + C
  return __builtin_amdgcn_wmma_f32_16x16x32_bf16(false, a, false, b, (short)0, c, false, false);
}

__device__ __forceinline__ v16bf cat16(v8bf a, v8bf b) {
  return __builtin_shufflevector(a, b, 0, 1, 2, 3, 4, 5, 6, 7, 8, 9, 10, 11, 12, 13, 14, 15);
}
// A-operand per-lane slots: p[0..15] (one 32B run, as two 16B loads)
__device__ __forceinline__ v16bf load_a(const __bf16* p) {
  return cat16(*(const v8bf*)p, *(const v8bf*)(p + 8));
}
// B-operand per-lane slots: p[0..15] (one 32B run, as two 16B loads)
__device__ __forceinline__ v16bf load_b(const __bf16* p) {
  return cat16(*(const v8bf*)p, *(const v8bf*)(p + 8));
}

__global__ void convert_weights_kernel(const float* __restrict__ Wq,
                                       const float* __restrict__ Wk,
                                       __bf16* __restrict__ wq16,
                                       __bf16* __restrict__ wk16) {
  int i = blockIdx.x * blockDim.x + threadIdx.x;
  if (i < CC * CC) {
    wq16[i] = (__bf16)Wq[i];
    wk16[i] = (__bf16)Wk[i];
  }
}

__global__ __launch_bounds__(NTHREADS, 1) void attention1d_kernel(
    const float* __restrict__ feature, const float* __restrict__ position,
    const __bf16* __restrict__ wq16, const float* __restrict__ bq,
    const __bf16* __restrict__ wk16, const float* __restrict__ bk,
    float* __restrict__ out) {
  extern __shared__ char smem_raw[];
  __bf16* fv = (__bf16*)smem_raw;    // [CC][FP]  values row-major, zero halo (out B-operand)
  __bf16* fvT = fv + CC * FP;        // [WW][CP]  values x-major (proj B-operand)
  __bf16* kT = fvT + WW * CP;        // [WP][CP]  keys x-major w/ halo rows (score B-operand)
  __bf16* qT = kT + WP * CP;         // [WW][CP]  queries x-major, pre-scaled (score A-operand)
  __bf16* PT = qT + WW * CP;         // [NWAVES][16][PP] attn weights x-major (out A-operand)

  const int bh = blockIdx.x;
  const int b = bh >> 6;   // / HH
  const int h = bh & 63;   // % HH
  const int tid = threadIdx.x;
  const int wv = tid >> 5;
  const int lane = tid & 31;
  const int n = lane & 15;
  const int hi = lane >> 4;

  const size_t plane = (size_t)HH * WW;
  const float* fbase = feature + (size_t)b * CC * plane + (size_t)h * WW;
  const float* pbase = position + (size_t)b * CC * plane + (size_t)h * WW;

  // ---- Phase 0: f = feature + position -> bf16, both layouts; zero halos
  for (int idx = tid; idx < CC * (WW / 4); idx += NTHREADS) {
    const int c = idx >> 6;          // / (WW/4)
    const int w = (idx & 63) << 2;   // 4 elements per thread
    const float4 f4 = *(const float4*)(fbase + (size_t)c * plane + w);
    const float4 p4 = *(const float4*)(pbase + (size_t)c * plane + w);
    const __bf16 v0 = (__bf16)(f4.x + p4.x);
    const __bf16 v1 = (__bf16)(f4.y + p4.y);
    const __bf16 v2 = (__bf16)(f4.z + p4.z);
    const __bf16 v3 = (__bf16)(f4.w + p4.w);
    __bf16* fr = fv + c * FP + RR + w;
    fr[0] = v0; fr[1] = v1; fr[2] = v2; fr[3] = v3;
    fvT[(w + 0) * CP + c] = v0;
    fvT[(w + 1) * CP + c] = v1;
    fvT[(w + 2) * CP + c] = v2;
    fvT[(w + 3) * CP + c] = v3;
  }
  for (int idx = tid; idx < CC * 2 * RR; idx += NTHREADS) {  // fv halo cols
    const int c = idx >> 4;
    const int j = idx & 15;
    const int p = (j < RR) ? j : (WW + RR + (j - RR));
    fv[c * FP + p] = (__bf16)0.0f;
  }
  for (int idx = tid; idx < 2 * RR * CC; idx += NTHREADS) {  // kT halo rows
    const int r = idx >> 7;
    const int c = idx & 127;
    const int row = (r < RR) ? r : (WW + RR + (r - RR));
    kT[row * CP + c] = (__bf16)0.0f;
  }
  __syncthreads();

  const float scale = 0.08838834764831845f;  // 1/sqrt(128)

  // ---- Phase A: Q/K projections. D(16o x 16x) = W(16o x 32c) * fvT-as-B(32c x 16x)
  for (int pass = 0; pass < 2; ++pass) {
    const __bf16* wmat = pass ? wk16 : wq16;
    const float* bias = pass ? bk : bq;
    for (int t = wv; t < 128; t += NWAVES) {
      const int o0 = (t >> 4) << 4;
      const int x0 = (t & 15) << 4;
      const __bf16* arowb = wmat + (size_t)(o0 + n) * CC + hi * 16;
      const __bf16* browb = fvT + (size_t)(x0 + n) * CP + hi * 16;
      v8f acc = {};
#pragma unroll
      for (int c0 = 0; c0 < CC; c0 += 32) {
        acc = wmma_bf16(load_a(arowb + c0), load_b(browb + c0), acc);
      }
      const int ob = o0 + 8 * hi;  // D: lane -> col x0+n ; reg r -> row ob+r
      if (pass == 0) {
        __bf16* qrow = qT + (size_t)(x0 + n) * CP + ob;
#pragma unroll
        for (int r = 0; r < 8; ++r) qrow[r] = (__bf16)((acc[r] + bias[ob + r]) * scale);
      } else {
        __bf16* krow = kT + (size_t)(RR + x0 + n) * CP + ob;
#pragma unroll
        for (int r = 0; r < 8; ++r) krow[r] = (__bf16)(acc[r] + bias[ob + r]);
      }
    }
  }
  __syncthreads();

  // ---- Phase B: banded attention, 2 x-blocks per wave
  __bf16* PTw = PT + wv * (16 * PP);
  for (int xb = wv; xb < WW / 16; xb += NWAVES) {
    const int x0 = xb << 4;
    // scores: S(16x x 32nc) = qT-as-A * kT-as-B ; padded key col for (m, j) = x0 + m + j
    const __bf16* qrowb = qT + (size_t)(x0 + n) * CP + hi * 16;
    const __bf16* k0row = kT + (size_t)(x0 + n) * CP + hi * 16;
    const __bf16* k1row = kT + (size_t)(x0 + 16 + n) * CP + hi * 16;
    v8f s0 = {}, s1 = {};
#pragma unroll
    for (int c0 = 0; c0 < CC; c0 += 32) {
      const v16bf qa = load_a(qrowb + c0);
      s0 = wmma_bf16(qa, load_b(k0row + c0), s0);
      s1 = wmma_bf16(qa, load_b(k1row + c0), s1);
    }
    // softmax over the 17-wide band; row m = r + 8*hi lives in one 16-lane half
#pragma unroll
    for (int r = 0; r < 8; ++r) {
      const int m = r + 8 * hi;
      const int nc0 = n;
      const int nc1 = 16 + n;
      const bool ok0 = (nc0 >= m) && (nc0 <= m + 16);
      const bool ok1 = (nc1 >= m) && (nc1 <= m + 16);
      float v0 = ok0 ? s0[r] : -3.0e38f;  // qT pre-scaled by 1/sqrt(C)
      float v1 = ok1 ? s1[r] : -3.0e38f;
      float mx = fmaxf(v0, v1);
#pragma unroll
      for (int off = 1; off < 16; off <<= 1) mx = fmaxf(mx, __shfl_xor(mx, off, 32));
      const float e0 = ok0 ? __expf(v0 - mx) : 0.0f;
      const float e1 = ok1 ? __expf(v1 - mx) : 0.0f;
      float sum = e0 + e1;
#pragma unroll
      for (int off = 1; off < 16; off <<= 1) sum += __shfl_xor(sum, off, 32);
      const float inv = 1.0f / sum;
      PTw[m * PP + nc0] = (__bf16)(e0 * inv);  // x-major for out A-operand
      PTw[m * PP + nc1] = (__bf16)(e1 * inv);
    }
    // out: O^T(16x x 16c) = PT-as-A(16x x 32nc) * fv-as-B(32nc x 16c)
    const v16bf pa = load_a(PTw + n * PP + hi * 16);
    float* obase = out + (size_t)b * CC * plane + (size_t)h * WW + x0 + 8 * hi;
#pragma unroll
    for (int ct = 0; ct < CC / 16; ++ct) {
      const int c0 = ct << 4;
      const v16bf vb = load_b(fv + (size_t)(c0 + n) * FP + x0 + hi * 16);
      v8f o = {};
      o = wmma_bf16(pa, vb, o);
      // D: lane -> channel c0+n ; reg r -> x = x0 + 8*hi + r (contiguous floats)
      float* optr = obase + (size_t)(c0 + n) * plane;
      *(float4*)(optr) = make_float4(o[0], o[1], o[2], o[3]);
      *(float4*)(optr + 4) = make_float4(o[4], o[5], o[6], o[7]);
    }
  }
}

extern "C" void kernel_launch(void* const* d_in, const int* in_sizes, int n_in,
                              void* d_out, int out_size, void* d_ws, size_t ws_size,
                              hipStream_t stream) {
  const float* feature = (const float*)d_in[0];
  const float* position = (const float*)d_in[1];
  const float* Wq = (const float*)d_in[2];
  const float* bq = (const float*)d_in[3];
  const float* Wk = (const float*)d_in[4];
  const float* bk = (const float*)d_in[5];
  float* out = (float*)d_out;

  __bf16* wq16 = (__bf16*)d_ws;
  __bf16* wk16 = wq16 + CC * CC;

  convert_weights_kernel<<<(CC * CC + 255) / 256, 256, 0, stream>>>(Wq, Wk, wq16, wk16);

  const size_t smem =
      (size_t)(CC * FP + WW * CP + WP * CP + WW * CP + NWAVES * 16 * PP) * sizeof(__bf16);
  // = 295,168 bytes; fits the 320 KB CDNA5 WGP LDS
  (void)hipFuncSetAttribute((const void*)attention1d_kernel,
                            hipFuncAttributeMaxDynamicSharedMemorySize, (int)smem);
  attention1d_kernel<<<BB * HH, NTHREADS, smem, stream>>>(feature, position, wq16, bq,
                                                          wk16, bk, out);
}
